// MultiHeadAttentionQuantum_65481071401202
// MI455X (gfx1250) — matrix operand order repaired and checked by reading, not compiled, not run on hardware
//
#include <hip/hip_runtime.h>

typedef __attribute__((ext_vector_type(16))) _Float16 v16h;
typedef __attribute__((ext_vector_type(8)))  float    v8f;
typedef __attribute__((ext_vector_type(4)))  unsigned v4u;
typedef __attribute__((ext_vector_type(8)))  unsigned v8u;

#define EMBED  512
#define SEQ    2048
#define BATCH  8
#define TOKENS (BATCH * SEQ)   // 16384

// ---------------- CDNA5 async / tensor helpers -------------------------------

__device__ __forceinline__ void async_b128(unsigned lds_addr, unsigned goff,
                                           unsigned long long sbase) {
  // GVS form: vdst = LDS byte address (VGPR), vaddr = 32-bit offset (VGPR),
  // saddr = 64-bit base (SGPR pair). Tracked with ASYNCcnt.
  asm volatile("global_load_async_to_lds_b128 %0, %1, %2"
               :: "v"(lds_addr), "v"(goff), "s"(sbase)
               : "memory");
}

__device__ __forceinline__ void wait_async0() {
  asm volatile("s_wait_asynccnt 0x0" ::: "memory");
}

// Tensor Data Mover: 2-D tile load, descriptor groups 0 (4 SGPRs) and
// 1 (8 SGPRs); groups 2/3 omitted (NULL) for tensors up to 2D. Tracked with
// TENSORcnt. EXEC ignored -> issue from a single wave only.
__device__ __forceinline__ void tdm_load(v4u g0, v8u g1) {
  asm volatile("tensor_load_to_lds %0, %1"
               :: "s"(g0), "s"(g1)
               : "memory");
}

// D# group 0: [1:0]=count(1), [63:32]=lds_addr, [120:64]=global_addr,
// [127:126]=type(2)
__device__ __forceinline__ v4u tdm_g0(unsigned lds_addr,
                                      unsigned long long gaddr) {
  v4u g0;
  g0[0] = 1u;
  g0[1] = lds_addr;
  g0[2] = (unsigned)gaddr;
  g0[3] = ((unsigned)(gaddr >> 32) & 0x01FFFFFFu) | (2u << 30);
  return g0;
}

// D# group 1: wg_mask=0, data_size=1 (2 bytes), tensor_dim0=512,
// tensor_dim1=rows, tile_dim0=32 (K), tile_dim1=128 (rows), tile_dim2=0,
// tensor_dim0_stride=512, dim1_stride unused.
__device__ __forceinline__ v8u tdm_g1(unsigned tensorRows) {
  v8u g1;
  g1[0] = (1u << 16);                                   // data_size = 2B
  g1[1] = ((unsigned)EMBED & 0xFFFFu) << 16;            // tensor_dim0 lo16
  g1[2] = ((unsigned)EMBED >> 16) | ((tensorRows & 0xFFFFu) << 16);
  g1[3] = (tensorRows >> 16) | (32u << 16);             // tile_dim0 = 32
  g1[4] = 128u;                                         // tile_dim1 = 128
  g1[5] = (unsigned)EMBED;                              // dim0_stride = 512
  g1[6] = 0u;
  g1[7] = 0u;
  return g1;
}

// ---------------- Kernel 1: quantum projection + per-token head attention ---
// One wave per token. x tile staged LDS via async-copy; proj = cos(x+theta)
// is 64x8 P; softmax(P P^T/sqrt(8)) P written into the scrambled Z layout:
//   Z[b, h*32 + (s>>6), (s&63)*8 + d] = attn[b,s,h,d]   (f16)

__global__ __launch_bounds__(256)
void qattn_proj_kernel(const float* __restrict__ x,
                       const float* __restrict__ theta,
                       _Float16* __restrict__ Zh) {
  __shared__ float xs[8][512];     // raw x staging (async-to-LDS)
  __shared__ float proj[8][512];   // cos(x + theta)
  const int wave = threadIdx.x >> 5;
  const int lane = threadIdx.x & 31;
  const int tok  = blockIdx.x * 8 + wave;

  // async-copy this block's 8 tokens of x; per-wave offset folded into voffset
  const unsigned ldsX = (unsigned)(size_t)(&xs[0][0]);
  const unsigned long long xb =
      (unsigned long long)(x + (size_t)blockIdx.x * 8 * EMBED);
  #pragma unroll
  for (int j = 0; j < 4; ++j) {
    const unsigned off = (unsigned)(wave * 2048 + j * 512 + lane * 16);
    async_b128(ldsX + off, off, xb);
  }
  wait_async0();   // intra-wave producer/consumer: no barrier needed

  #pragma unroll
  for (int i = 0; i < 16; ++i) {
    const int idx = i * 32 + lane;
    proj[wave][idx] = __cosf(xs[wave][idx] + theta[idx & 7]);
  }
  __syncthreads();

  const float scale = 0.35355339059327373f;   // 1/sqrt(8)
  const int s  = tok & (SEQ - 1);
  const int bb = tok >> 11;
  const float* P = proj[wave];

  #pragma unroll
  for (int r = 0; r < 2; ++r) {
    const int h = lane + r * 32;              // two head-rows per lane
    float ph[8];
    #pragma unroll
    for (int d = 0; d < 8; ++d) ph[d] = P[h * 8 + d];

    float m = -1e30f;
    for (int g = 0; g < 64; ++g) {
      float sd = 0.f;
      #pragma unroll
      for (int d = 0; d < 8; ++d) sd = fmaf(ph[d], P[g * 8 + d], sd);
      m = fmaxf(m, sd * scale);
    }
    float sum = 0.f;
    float acc[8] = {0.f, 0.f, 0.f, 0.f, 0.f, 0.f, 0.f, 0.f};
    for (int g = 0; g < 64; ++g) {
      float sd = 0.f;
      #pragma unroll
      for (int d = 0; d < 8; ++d) sd = fmaf(ph[d], P[g * 8 + d], sd);
      const float e = __expf(sd * scale - m);
      sum += e;
      #pragma unroll
      for (int d = 0; d < 8; ++d) acc[d] = fmaf(e, P[g * 8 + d], acc[d]);
    }
    const float inv = __frcp_rn(sum);

    union { _Float16 hv[8]; v4u q; } u;
    #pragma unroll
    for (int d = 0; d < 8; ++d) u.hv[d] = (_Float16)(acc[d] * inv);
    const size_t zi =
        ((size_t)bb * SEQ + (size_t)h * 32 + (size_t)(s >> 6)) * EMBED +
        (size_t)(s & 63) * 8;                  // 16B aligned
    *(v4u*)(Zh + zi) = u.q;
  }
}

// ---------------- Kernel 0: W_combine f32 -> f16 -----------------------------

__global__ __launch_bounds__(256)
void cvt_w_kernel(const float* __restrict__ W, _Float16* __restrict__ Wh) {
  const int i = blockIdx.x * 256 + threadIdx.x;
  Wh[i] = (_Float16)W[i];
}

// ---------------- Kernel 2: out = Z @ W^T + bias  (16384 x 512 x 512) --------
// f16 in, f32 accumulate via v_wmma_f32_16x16x32_f16.
// Block tile 128x128, K-step 32, 8 waves (4M x 2N), wave tile 32x64.
// Double-buffered LDS fed by the Tensor Data Mover (TENSORcnt), DMA driven
// by wave 0 while all 8 waves run WMMA.

#define BM 128
#define BN 128
#define BK 32
#define KSTEPS       (EMBED / BK)       // 16
#define TILE_HALVES  (BM * BK)          // 4096 halves = 8KB
#define STAGE_HALVES (2 * TILE_HALVES)  // A+B per buffer

union FragAB { v16h h; v4u q[2]; };

__global__ __launch_bounds__(256)
void gemm_wmma_kernel(const _Float16* __restrict__ Zh,
                      const _Float16* __restrict__ Wh,
                      const float* __restrict__ bias,
                      float* __restrict__ out) {
  __shared__ _Float16 smem[2 * STAGE_HALVES];   // 32KB, double buffered

  const int tid    = threadIdx.x;
  const int lane   = tid & 31;
  const int waveId = tid >> 5;
  const int mTile  = blockIdx.x;
  const int nTile  = blockIdx.y;
  const int m0 = (waveId & 3) * 32;
  const int n0 = (waveId >> 2) * 64;

  const unsigned ldsBase = (unsigned)(size_t)(&smem[0]);
  const unsigned long long AbaseB =
      (unsigned long long)(Zh + (size_t)mTile * BM * EMBED);
  const unsigned long long BbaseB =
      (unsigned long long)(Wh + (size_t)nTile * BN * EMBED);

  const v8u g1A = tdm_g1((unsigned)TOKENS);   // A tensor: 16384 rows
  const v8u g1B = tdm_g1((unsigned)EMBED);    // B tensor: 512 rows

  auto prefetch = [&](int ks, int buf) {      // wave 0 only (EXEC ignored)
    const unsigned bufOff = (unsigned)buf * (STAGE_HALVES * 2);
    const unsigned long long kByte = (unsigned long long)(ks * BK * 2);
    tdm_load(tdm_g0(ldsBase + bufOff, AbaseB + kByte), g1A);
    tdm_load(tdm_g0(ldsBase + bufOff + TILE_HALVES * 2, BbaseB + kByte), g1B);
  };

  const v8f zero8 = {0.f, 0.f, 0.f, 0.f, 0.f, 0.f, 0.f, 0.f};
  v8f acc[2][4];
  #pragma unroll
  for (int mi = 0; mi < 2; ++mi)
    #pragma unroll
    for (int ni = 0; ni < 4; ++ni) acc[mi][ni] = zero8;

  if (waveId == 0) prefetch(0, 0);

  // ISA 7.12.2 fragment sub-offsets (halves):
  // A 16x32: lanes 0-15 hold K{0-7,16-23}; lanes 16-31 hold K{8-15,24-31}
  // B 32x16: lanes 0-15 hold K{0-7,16-23} of col n; lanes 16-31 hold K{8-15,24-31}
  const int aoff = (lane & 16) ? 8 : 0;
  const int boff = (lane & 16) ? 8 : 0;

  for (int ks = 0; ks < KSTEPS; ++ks) {
    if (waveId == 0) __builtin_amdgcn_s_wait_tensorcnt(0);  // DMA of stage ks
    __syncthreads();                                        // publish to WGP
    if (waveId == 0 && ks + 1 < KSTEPS)
      prefetch(ks + 1, (ks + 1) & 1);                       // overlap compute

    const _Float16* sA = smem + (ks & 1) * STAGE_HALVES;
    const _Float16* sB = sA + TILE_HALVES;

    FragAB a[2], b[4];
    #pragma unroll
    for (int mi = 0; mi < 2; ++mi) {
      const v4u* p = (const v4u*)(sA + (m0 + mi * 16 + (lane & 15)) * BK + aoff);
      a[mi].q[0] = p[0];
      a[mi].q[1] = p[2];        // +16 halves
    }
    #pragma unroll
    for (int ni = 0; ni < 4; ++ni) {
      const v4u* p = (const v4u*)(sB + (n0 + ni * 16 + (lane & 15)) * BK + boff);
      b[ni].q[0] = p[0];
      b[ni].q[1] = p[2];        // +16 halves
    }
    #pragma unroll
    for (int mi = 0; mi < 2; ++mi)
      #pragma unroll
      for (int ni = 0; ni < 4; ++ni)
        acc[mi][ni] = __builtin_amdgcn_wmma_f32_16x16x32_f16(
            false, a[mi].h, false, b[ni].h, (short)0, acc[mi][ni],
            false, false);
  }

  // Epilogue: C layout — VGPR r: lanes 0-15 -> M=r, lanes 16-31 -> M=r+8,
  // N = lane&15. Add bias, store f32.
  const int rowAdd = (lane & 16) ? 8 : 0;
  #pragma unroll
  for (int ni = 0; ni < 4; ++ni) {
    const int gn = nTile * BN + n0 + ni * 16 + (lane & 15);
    const float bn = bias[gn];
    #pragma unroll
    for (int mi = 0; mi < 2; ++mi) {
      #pragma unroll
      for (int r = 0; r < 8; ++r) {
        const int gm = mTile * BM + m0 + mi * 16 + rowAdd + r;
        out[(size_t)gm * EMBED + gn] = acc[mi][ni][r] + bn;
      }
    }
  }
}

// ---------------- launch -----------------------------------------------------

extern "C" void kernel_launch(void* const* d_in, const int* in_sizes, int n_in,
                              void* d_out, int out_size, void* d_ws, size_t ws_size,
                              hipStream_t stream) {
  (void)in_sizes; (void)n_in; (void)out_size; (void)ws_size;
  const float* x     = (const float*)d_in[0];
  const float* theta = (const float*)d_in[1];
  const float* W     = (const float*)d_in[2];
  const float* bias  = (const float*)d_in[3];
  float* out = (float*)d_out;

  // workspace: Zh (16384*512 halves = 16MB) then Wh (512*512 halves = 0.5MB)
  _Float16* Zh = (_Float16*)d_ws;
  _Float16* Wh = Zh + (size_t)TOKENS * EMBED;

  qattn_proj_kernel<<<TOKENS / 8, 256, 0, stream>>>(x, theta, Zh);
  cvt_w_kernel<<<(EMBED * EMBED) / 256, 256, 0, stream>>>(W, Wh);

  dim3 grid(TOKENS / BM, EMBED / BN);   // 128 x 4 = 512 workgroups
  gemm_wmma_kernel<<<grid, 256, 0, stream>>>(Zh, Wh, bias, out);
}
